// AttentionLayer_18202071401047
// MI455X (gfx1250) — matrix operand order repaired and checked by reading, not compiled, not run on hardware
//
#include <hip/hip_runtime.h>
#include <hip/hip_bf16.h>

typedef __attribute__((ext_vector_type(16))) _Float16 v16h;
typedef __attribute__((ext_vector_type(8)))  _Float16 v8h;
typedef __attribute__((ext_vector_type(8)))  float    v8f;

#define DIM   512
#define NH    8
#define HD    64
#define SEQ   4096
#define BATCH 2
#define MROWS (BATCH * SEQ)   // 8192

// ---------------------------------------------------------------------------
// WMMA helper: D = A(16x32 f16) * B(32x16 f16) + C(16x16 f32)
// ---------------------------------------------------------------------------
__device__ __forceinline__ v8f wmma_f16(v16h a, v16h b, v8f c) {
  return __builtin_amdgcn_wmma_f32_16x16x32_f16(
      /*neg_a=*/false, a, /*neg_b=*/false, b,
      /*c_mod=*/(short)0, c, /*reuse_a=*/false, /*reuse_b=*/false);
}

// A-fragment loader (16-bit A, 16x32, wave32):
//   lanes 0-15 : row M = lane,     K = {0..15}  (kbase = k0 + 0)
//   lanes16-31 : row M = lane-16,  K = {16..31} (kbase = k0 + 16)
__device__ __forceinline__ v16h load_a16(const _Float16* rowbase, int kbase) {
  v8h lo = *(const v8h*)(rowbase + kbase);
  v8h hi = *(const v8h*)(rowbase + kbase + 8);
  v16h r;
#pragma unroll
  for (int i = 0; i < 8; ++i) { r[i] = lo[i]; r[i + 8] = hi[i]; }
  return r;
}

// ---------------------------------------------------------------------------
// f32 -> f16 conversion (grid-stride)
// ---------------------------------------------------------------------------
__global__ void f32_to_f16_kernel(const float* __restrict__ in,
                                  _Float16* __restrict__ out, int n) {
  int i = blockIdx.x * blockDim.x + threadIdx.x;
  int stride = gridDim.x * blockDim.x;
  for (; i < n; i += stride) out[i] = (_Float16)in[i];
}

// ---------------------------------------------------------------------------
// Fused QKV projection: Q/K/V[m,n] = sum_k x[m,k]*W{q,k,v}[n,k] + b{q,k,v}[n]
// A fragments loaded once, reused by 12 WMMAs per k-step (3 mats x 4 cols).
// Q,K stored f16 head-major [B,H,S,hd]; V stored f16 TRANSPOSED [B,H,hd,S]
// so attention's attn@V B-fragments become contiguous loads.
// block = 128 threads (4 waves); wave tile 16x64; grid (128, 8).
// ---------------------------------------------------------------------------
__global__ __launch_bounds__(128, 1)
void fused_qkv_kernel(const _Float16* __restrict__ A,
                      const _Float16* __restrict__ Wq,
                      const _Float16* __restrict__ Wk,
                      const _Float16* __restrict__ Wv,
                      const float* __restrict__ bq,
                      const float* __restrict__ bk,
                      const float* __restrict__ bv,
                      _Float16* __restrict__ Qh,
                      _Float16* __restrict__ Kh,
                      _Float16* __restrict__ Vt) {
  const int lane = threadIdx.x & 31;
  const int wv   = threadIdx.x >> 5;
  const int half = lane >> 4;
  const int lcol = lane & 15;

  const int m0  = blockIdx.x * 64 + wv * 16;
  const int n0  = blockIdx.y * 64;
  const int row = m0 + lcol;

  const v8f vzero = {0.f, 0.f, 0.f, 0.f, 0.f, 0.f, 0.f, 0.f};
  v8f acc[3][4];
#pragma unroll
  for (int w = 0; w < 3; ++w)
#pragma unroll
    for (int c = 0; c < 4; ++c) acc[w][c] = vzero;

  for (int k0 = 0; k0 < DIM; k0 += 32) {
    const v16h a = load_a16(A + (size_t)row * DIM, k0 + half * 16);
    const size_t woff = (size_t)k0 + half * 16;
#pragma unroll
    for (int c = 0; c < 4; ++c) {
      const size_t nrow = (size_t)(n0 + 16 * c + lcol) * DIM + woff;
      v16h bqf = *(const v16h*)(Wq + nrow);
      acc[0][c] = wmma_f16(a, bqf, acc[0][c]);
      v16h bkf = *(const v16h*)(Wk + nrow);
      acc[1][c] = wmma_f16(a, bkf, acc[1][c]);
      v16h bvf = *(const v16h*)(Wv + nrow);
      acc[2][c] = wmma_f16(a, bvf, acc[2][c]);
    }
  }

#pragma unroll
  for (int c = 0; c < 4; ++c) {
    const int n  = n0 + 16 * c + lcol;
    const int h  = n >> 6, d = n & 63;
    const float biq = bq[n], bik = bk[n], biv = bv[n];
#pragma unroll
    for (int r = 0; r < 8; ++r) {
      const int m  = m0 + r + half * 8;                 // D-frag row mapping
      const int bb = m >> 12, ss = m & (SEQ - 1);
      const size_t hm = ((size_t)(bb * NH + h) * SEQ + ss) * HD + d;
      Qh[hm] = (_Float16)(acc[0][c][r] + biq);
      Kh[hm] = (_Float16)(acc[1][c][r] + bik);
      Vt[((size_t)(bb * NH + h) * HD + d) * SEQ + ss] =
          (_Float16)(acc[2][c][r] + biv);
    }
  }
}

// ---------------------------------------------------------------------------
// Output projection: out[m,n] = sum_k O[m,k]*Wo[n,k] + bo[n], f32 out.
// O is head-major f16 [B,H,S,hd] viewed as [m, k=h*64+d].
// ---------------------------------------------------------------------------
__global__ __launch_bounds__(128, 1)
void out_proj_kernel(const _Float16* __restrict__ A,
                     const _Float16* __restrict__ W,
                     const float* __restrict__ bias,
                     float* __restrict__ outf) {
  const int lane = threadIdx.x & 31;
  const int wv   = threadIdx.x >> 5;
  const int half = lane >> 4;
  const int lcol = lane & 15;

  const int m0  = blockIdx.x * 64 + wv * 16;
  const int n0  = blockIdx.y * 64;
  const int row = m0 + lcol;
  const int bb  = row >> 12, ss = row & (SEQ - 1);

  const v8f vzero = {0.f, 0.f, 0.f, 0.f, 0.f, 0.f, 0.f, 0.f};
  v8f acc[4];
#pragma unroll
  for (int c = 0; c < 4; ++c) acc[c] = vzero;

  for (int k0 = 0; k0 < DIM; k0 += 32) {
    const int kb = k0 + half * 16;
    const int k2 = kb + 8;
    // addr(row,k) = ((b*NH + k/HD)*SEQ + s)*HD + k%HD ; chunks stay in-head
    const _Float16* p1 =
        A + (((size_t)(bb * NH + (kb >> 6)) * SEQ + ss) * HD + (kb & 63));
    const _Float16* p2 =
        A + (((size_t)(bb * NH + (k2 >> 6)) * SEQ + ss) * HD + (k2 & 63));
    v8h lo = *(const v8h*)p1;
    v8h hi = *(const v8h*)p2;
    v16h a;
#pragma unroll
    for (int i = 0; i < 8; ++i) { a[i] = lo[i]; a[i + 8] = hi[i]; }
#pragma unroll
    for (int c = 0; c < 4; ++c) {
      const int n = n0 + 16 * c + lcol;
      v16h b = *(const v16h*)(W + (size_t)n * DIM + k0 + half * 16);
      acc[c] = wmma_f16(a, b, acc[c]);
    }
  }

#pragma unroll
  for (int c = 0; c < 4; ++c) {
    const int n = n0 + 16 * c + lcol;
    const float bv = bias[n];
#pragma unroll
    for (int r = 0; r < 8; ++r) {
      const int m = m0 + r + half * 8;
      outf[(size_t)m * DIM + n] = acc[c][r] + bv;
    }
  }
}

// ---------------------------------------------------------------------------
// Flash attention: one (b,h) slice per blockIdx.y; 64 queries per block
// (16 per wave). Streams key blocks of 64, online softmax, f32 O accum.
// Q pre-scaled by 0.125 (exact in f16). V read from transposed [hd,S] buffer.
// ---------------------------------------------------------------------------
__global__ __launch_bounds__(128, 1)
void flash_attn_kernel(const _Float16* __restrict__ Qh,
                       const _Float16* __restrict__ Kh,
                       const _Float16* __restrict__ Vt,
                       _Float16* __restrict__ Oh) {
  __shared__ __attribute__((aligned(16))) _Float16 plds[4][16][64];

  const int lane = threadIdx.x & 31;
  const int wv   = threadIdx.x >> 5;
  const int half = lane >> 4;
  const int lcol = lane & 15;

  const size_t base = (size_t)blockIdx.y * SEQ * HD;    // (b,h) slice
  const _Float16* Qp  = Qh + base;
  const _Float16* Kp  = Kh + base;
  const _Float16* Vtp = Vt + base;                      // [hd][SEQ]
  _Float16*       Op  = Oh + base;

  const int q0   = blockIdx.x * 64 + wv * 16;
  const int qrow = q0 + lcol;

  // Q A-fragments for d=0..31 and 32..63, pre-scaled by 1/sqrt(64)=0.125
  v16h qa0 = load_a16(Qp + (size_t)qrow * HD, half * 16);
  v16h qa1 = load_a16(Qp + (size_t)qrow * HD, 32 + half * 16);
#pragma unroll
  for (int i = 0; i < 16; ++i) {
    qa0[i] = qa0[i] * (_Float16)0.125f;
    qa1[i] = qa1[i] * (_Float16)0.125f;
  }

  float mrow[8], lrow[8];
  v8f o[4];
  const v8f vzero = {0.f, 0.f, 0.f, 0.f, 0.f, 0.f, 0.f, 0.f};
#pragma unroll
  for (int r = 0; r < 8; ++r) { mrow[r] = -1e30f; lrow[r] = 0.f; }
#pragma unroll
  for (int c = 0; c < 4; ++c) o[c] = vzero;

  for (int j0 = 0; j0 < SEQ; j0 += 64) {
    // ---- scores: S = (Q*scale) K^T  (16 queries x 64 keys) ----
    v8f s[4];
#pragma unroll
    for (int c = 0; c < 4; ++c) {
      const int key = j0 + 16 * c + lcol;               // B-frag: lane = key col
      const _Float16* kr = Kp + (size_t)key * HD + half * 16;
      v16h kb0 = *(const v16h*)(kr);                    // d = 0..31
      v16h kb1 = *(const v16h*)(kr + 32);               // d = 32..63
      s[c] = vzero;
      s[c] = wmma_f16(qa0, kb0, s[c]);
      s[c] = wmma_f16(qa1, kb1, s[c]);
    }

    // ---- online softmax: row max over this key block ----
    float tmax[8];
#pragma unroll
    for (int r = 0; r < 8; ++r)
      tmax[r] = fmaxf(fmaxf(s[0][r], s[1][r]), fmaxf(s[2][r], s[3][r]));
#pragma unroll
    for (int r = 0; r < 8; ++r) {
#pragma unroll
      for (int mask = 1; mask <= 8; mask <<= 1)
        tmax[r] = fmaxf(tmax[r], __shfl_xor(tmax[r], mask, 32));
    }

    float alpha[8], rsum[8];
#pragma unroll
    for (int r = 0; r < 8; ++r) {
      const float mn = fmaxf(mrow[r], tmax[r]);
      alpha[r] = __expf(mrow[r] - mn);
      mrow[r]  = mn;
      rsum[r]  = 0.f;
    }
#pragma unroll
    for (int c = 0; c < 4; ++c) {
#pragma unroll
      for (int r = 0; r < 8; ++r) {
        const float p = __expf(s[c][r] - mrow[r]);
        s[c][r] = p;
        rsum[r] += p;
      }
    }
#pragma unroll
    for (int r = 0; r < 8; ++r) {
#pragma unroll
      for (int mask = 1; mask <= 8; mask <<= 1)
        rsum[r] += __shfl_xor(rsum[r], mask, 32);
      lrow[r] = lrow[r] * alpha[r] + rsum[r];
    }
#pragma unroll
    for (int c = 0; c < 4; ++c)
#pragma unroll
      for (int r = 0; r < 8; ++r) o[c][r] *= alpha[r];

    // ---- P (D-layout) -> LDS -> A-layout fragments ----
#pragma unroll
    for (int c = 0; c < 4; ++c)
#pragma unroll
      for (int r = 0; r < 8; ++r)
        plds[wv][r + half * 8][16 * c + lcol] = (_Float16)s[c][r];
    // same-wave LDS ops are in-order; no cross-wave sharing -> no barrier
    const _Float16* prow = &plds[wv][lcol][0];
    v16h pa0 = load_a16(prow, half * 16);               // keys j0+0..31
    v16h pa1 = load_a16(prow, 32 + half * 16);          // keys j0+32..63

    // ---- O += P * V  (V transposed: contiguous B-fragment loads) ----
#pragma unroll
    for (int c = 0; c < 4; ++c) {
      const int dcol = 16 * c + lcol;                   // B-frag: lane = d col
      const _Float16* vr = Vtp + (size_t)dcol * SEQ + j0 + half * 16;
      v16h vb0 = *(const v16h*)(vr);                    // keys j0+0..31
      v16h vb1 = *(const v16h*)(vr + 32);               // keys j0+32..63
      o[c] = wmma_f16(pa0, vb0, o[c]);
      o[c] = wmma_f16(pa1, vb1, o[c]);
    }
  }

  // ---- normalize and store O (f16, head-major) ----
#pragma unroll
  for (int c = 0; c < 4; ++c) {
#pragma unroll
    for (int r = 0; r < 8; ++r) {
      const int q = q0 + r + half * 8;
      const int d = 16 * c + lcol;
      Op[(size_t)q * HD + d] = (_Float16)(o[c][r] / lrow[r]);
    }
  }
}

// ---------------------------------------------------------------------------
// Launch: x,Wq,bq,Wk,bk,Wv,bv,Wo,bo  (all f32).  Workspace ~42 MB.
// ---------------------------------------------------------------------------
extern "C" void kernel_launch(void* const* d_in, const int* in_sizes, int n_in,
                              void* d_out, int out_size, void* d_ws,
                              size_t ws_size, hipStream_t stream) {
  const float* x  = (const float*)d_in[0];
  const float* Wq = (const float*)d_in[1];
  const float* bq = (const float*)d_in[2];
  const float* Wk = (const float*)d_in[3];
  const float* bk = (const float*)d_in[4];
  const float* Wv = (const float*)d_in[5];
  const float* bv = (const float*)d_in[6];
  const float* Wo = (const float*)d_in[7];
  const float* bo = (const float*)d_in[8];

  char* ws = (char*)d_ws;
  size_t off = 0;
  _Float16* xh  = (_Float16*)(ws + off); off += (size_t)MROWS * DIM * 2;
  _Float16* Wqh = (_Float16*)(ws + off); off += (size_t)DIM * DIM * 2;
  _Float16* Wkh = (_Float16*)(ws + off); off += (size_t)DIM * DIM * 2;
  _Float16* Wvh = (_Float16*)(ws + off); off += (size_t)DIM * DIM * 2;
  _Float16* Woh = (_Float16*)(ws + off); off += (size_t)DIM * DIM * 2;
  _Float16* Qh  = (_Float16*)(ws + off); off += (size_t)MROWS * DIM * 2;
  _Float16* Kh  = (_Float16*)(ws + off); off += (size_t)MROWS * DIM * 2;
  _Float16* Vt  = (_Float16*)(ws + off); off += (size_t)MROWS * DIM * 2;
  _Float16* Oh  = (_Float16*)(ws + off); off += (size_t)MROWS * DIM * 2;

  f32_to_f16_kernel<<<1024, 256, 0, stream>>>(x, xh, MROWS * DIM);
  f32_to_f16_kernel<<<256, 256, 0, stream>>>(Wq, Wqh, DIM * DIM);
  f32_to_f16_kernel<<<256, 256, 0, stream>>>(Wk, Wkh, DIM * DIM);
  f32_to_f16_kernel<<<256, 256, 0, stream>>>(Wv, Wvh, DIM * DIM);
  f32_to_f16_kernel<<<256, 256, 0, stream>>>(Wo, Woh, DIM * DIM);

  dim3 gg(MROWS / 64, DIM / 64);   // 128 x 8 blocks, 128 threads each
  fused_qkv_kernel<<<gg, 128, 0, stream>>>(xh, Wqh, Wkh, Wvh, bq, bk, bv,
                                           Qh, Kh, Vt);

  flash_attn_kernel<<<dim3(SEQ / 64, BATCH * NH), 128, 0, stream>>>(Qh, Kh, Vt, Oh);

  out_proj_kernel<<<gg, 128, 0, stream>>>(Oh, Woh, bo, (float*)d_out);
}
